// EELP_83227876262319
// MI455X (gfx1250) — compile-verified, matrix operand order checked by reading, not run on hardware
//
#include <hip/hip_runtime.h>
#include <hip/hip_bf16.h>

typedef __attribute__((ext_vector_type(16))) _Float16 v16h;
typedef __attribute__((ext_vector_type(8)))  float    v8f;

#define NN    20000
#define EE    320000
#define PP    50000
#define LL    20
#define HD    64

// ---------------- WMMA helpers (16x16x32 f16 -> f32) ----------------

__device__ __forceinline__ int kbase_of(int v, int kh) {
  // 16-bit A-matrix 16x32 layout (ISA 7.12.2): VGPR v holds K pair
  return (v < 4) ? ((kh ? 8 : 0) + 2 * v) : ((kh ? 24 : 16) + 2 * (v - 4));
}

// Load a 16x32 A tile (row-major f16 in LDS, row stride `ld`, k offset k0)
__device__ __forceinline__ v16h lds_load_A(const _Float16* tile, int ld, int k0) {
  int lane = threadIdx.x & 31;
  int m = lane & 15, kh = (lane >> 4) & 1;
  v16h a;
#pragma unroll
  for (int v = 0; v < 8; ++v) {
    int kb = k0 + kbase_of(v, kh);
    a[2 * v]     = tile[m * ld + kb];
    a[2 * v + 1] = tile[m * ld + kb + 1];
  }
  return a;
}

// B tiles pre-packed per-lane: lane's 16 halves contiguous (32B) ->
// two global_load_b128, fully L2 resident.
__device__ __forceinline__ v16h load_B_packed(const _Float16* pk, int ktiles, int kt, int nt) {
  int lane = threadIdx.x & 31;
  const _Float16* p = pk + ((size_t)((nt * ktiles + kt) * 32 + lane) << 4);
  return *(const v16h*)p;
}

__device__ __forceinline__ v8f wmma16(const v16h& a, const v16h& b, v8f c) {
  return __builtin_amdgcn_wmma_f32_16x16x32_f16(false, a, false, b, (short)0, c, false, false);
}

// Async global->LDS copy of one 16B chunk (CDNA5 GLOBAL_LOAD_ASYNC_TO_LDS_B128,
// tracked by ASYNCcnt). lds_off is the wave-relative LDS byte address.
__device__ __forceinline__ void async_copy_b128(unsigned lds_off, const void* gaddr) {
  asm volatile("global_load_async_to_lds_b128 %0, %1, off"
               :: "v"(lds_off), "v"(gaddr) : "memory");
}
__device__ __forceinline__ void async_wait0() {
  asm volatile("s_wait_asynccnt 0" ::: "memory");
}

// ---------------- weight packing ----------------

__global__ void k_pack_weight(const float* __restrict__ W, _Float16* __restrict__ out,
                              int K, int Nw) {
  int idx = blockIdx.x * 256 + threadIdx.x;
  int total = K * Nw;
  if (idx >= total) return;
  int i    = idx & 15;
  int lane = (idx >> 4) & 31;
  int t    = idx >> 9;                 // linear tile = nt*ktiles + kt
  int ktiles = K >> 5;
  int kt = t % ktiles, nt = t / ktiles;
  int v = i >> 1, pr = i & 1, kh = lane >> 4, n = lane & 15;
  int k  = kt * 32 + kbase_of(v, kh) + pr;
  int nn = nt * 16 + n;
  out[idx] = (_Float16)W[k * Nw + nn];
}

__global__ void k_prep_small(const float* __restrict__ Omega, const float* __restrict__ WsRaw,
                             const float* __restrict__ Wfc2, const float* __restrict__ Wfv,
                             const float* __restrict__ Wp2,
                             float* OmegaAs, float* Wsym, float* Wfc2pad, float* Wfvpad,
                             float* Wp2pad) {
  int i = blockIdx.x * 256 + threadIdx.x;
  if (i < HD * HD) {
    int r = i >> 6, c = i & 63;
    OmegaAs[i] = Omega[i] - Omega[c * HD + r];
    Wsym[i]    = 0.5f * (WsRaw[i] + WsRaw[c * HD + r]);
  }
  if (i < 128 * 16) {
    int k = i >> 4, c = i & 15;
    Wfc2pad[i] = (c < 2) ? Wfc2[k * 2 + c] : 0.0f;
    Wfvpad[i]  = (c == 0) ? Wfv[k] : 0.0f;
  }
  if (i < 64 * 16) {
    int k = i >> 4, c = i & 15;
    Wp2pad[i] = (c == 0) ? Wp2[k] : 0.0f;
  }
}

// ---------------- graph normalization ----------------

__global__ void k_init(float* deg, float* ts) {
  int i = blockIdx.x * 256 + threadIdx.x;
  if (i < NN) deg[i] = 1.0f;   // self loop contributes 1
  if (i < PP) ts[i] = 0.0f;
}

__global__ void k_deg(const int* __restrict__ ei, float* deg) {
  int e = blockIdx.x * 256 + threadIdx.x;
  if (e < EE) atomicAdd(&deg[ei[EE + e]], 1.0f);
}

__global__ void k_dinv(const float* __restrict__ deg, float* dinv) {
  int i = blockIdx.x * 256 + threadIdx.x;
  if (i < NN) dinv[i] = rsqrtf(deg[i]);   // deg >= 1 always
}

__global__ void k_norm(const int* __restrict__ ei, const float* __restrict__ dinv,
                       float* norm) {
  int e = blockIdx.x * 256 + threadIdx.x;
  if (e >= EE + NN) return;
  int r = (e < EE) ? ei[e] : (e - EE);
  int c = (e < EE) ? ei[EE + e] : (e - EE);
  norm[e] = dinv[r] * dinv[c];
}

// ---------------- encoder: H = relu(x @ W_enc), keep f32 + f16 mirror ----------------

__global__ void k_encode(const float* __restrict__ x, const _Float16* __restrict__ enc_pk,
                         float* __restrict__ Hm, _Float16* __restrict__ Hh) {
  __shared__ __align__(16) _Float16 sh_x[4][16 * 128];
  int wid = threadIdx.x >> 5, lane = threadIdx.x & 31;
  int rt = blockIdx.x * 4 + wid;
  bool act = rt < (NN / 16);
  if (act) {
    for (int idx = lane; idx < 16 * 128; idx += 32) {
      int m = idx >> 7, f = idx & 127;
      sh_x[wid][idx] = (_Float16)x[(size_t)(rt * 16 + m) * 128 + f];
    }
  }
  __syncthreads();
  if (act) {
    int n = lane & 15, m0 = (lane >> 4) * 8;
#pragma unroll
    for (int nt = 0; nt < 4; ++nt) {
      v8f acc = {0.f, 0.f, 0.f, 0.f, 0.f, 0.f, 0.f, 0.f};
#pragma unroll
      for (int kt = 0; kt < 4; ++kt)
        acc = wmma16(lds_load_A(&sh_x[wid][0], 128, kt * 32),
                     load_B_packed(enc_pk, 4, kt, nt), acc);
      size_t base = (size_t)(rt * 16) * HD + nt * 16;
#pragma unroll
      for (int i = 0; i < 8; ++i) {
        float hv = fmaxf(acc[i], 0.0f);
        Hm[base + (size_t)(m0 + i) * HD + n] = hv;
        Hh[base + (size_t)(m0 + i) * HD + n] = (_Float16)hv;
      }
    }
  }
}

// ---------------- r init ----------------

__global__ void k_rinit(const float* __restrict__ Hm, const int* __restrict__ pairs,
                        float* __restrict__ r) {
  int t = blockIdx.x * 256 + threadIdx.x;
  if (t >= PP * 128) return;
  int p = t >> 7, f = t & 127;
  int node = (f < 64) ? pairs[2 * p] : pairs[2 * p + 1];
  r[t] = Hm[node * HD + (f & 63)];
}

// ---------------- per-step node GEMMs: anti = -relu(H@Omega_as), M = H@W_s ----------------

__global__ void k_HW(const float* __restrict__ Hm, const _Float16* __restrict__ om_pk,
                     const _Float16* __restrict__ ws_pk, float* __restrict__ anti,
                     float* __restrict__ Msum) {
  __shared__ __align__(16) _Float16 sh_H[8][16 * 64];
  int wid = threadIdx.x >> 5, lane = threadIdx.x & 31;
  int rt = blockIdx.x * 8 + wid;
  bool act = rt < (NN / 16);
  if (act) {
    for (int idx = lane; idx < 16 * 64; idx += 32) {
      int m = idx >> 6, f = idx & 63;
      sh_H[wid][idx] = (_Float16)Hm[(size_t)(rt * 16 + m) * HD + f];
    }
  }
  __syncthreads();
  if (act) {
    int n = lane & 15, m0 = (lane >> 4) * 8;
#pragma unroll
    for (int nt = 0; nt < 4; ++nt) {
      v8f aA = {0.f, 0.f, 0.f, 0.f, 0.f, 0.f, 0.f, 0.f};
      v8f aS = {0.f, 0.f, 0.f, 0.f, 0.f, 0.f, 0.f, 0.f};
#pragma unroll
      for (int kt = 0; kt < 2; ++kt) {
        v16h a = lds_load_A(&sh_H[wid][0], 64, kt * 32);
        aA = wmma16(a, load_B_packed(om_pk, 2, kt, nt), aA);
        aS = wmma16(a, load_B_packed(ws_pk, 2, kt, nt), aS);
      }
      size_t base = (size_t)(rt * 16) * HD + nt * 16;
#pragma unroll
      for (int i = 0; i < 8; ++i) {
        anti[base + (size_t)(m0 + i) * HD + n] = -fmaxf(aA[i], 0.0f);
        Msum[base + (size_t)(m0 + i) * HD + n] = aS[i];
      }
    }
  }
}

__global__ void k_zero(float* buf, long long count) {
  long long i = (long long)blockIdx.x * 256 + threadIdx.x;
  if (i < count) buf[i] = 0.0f;
}

// ---------------- scatter: sym[col] += M[row]*norm (float4 per thread, L2 atomics) ----------

__global__ void k_scatter(const int* __restrict__ ei, const float* __restrict__ norm,
                          const float* __restrict__ Msum, float* __restrict__ sym) {
  int t = blockIdx.x * 256 + threadIdx.x;
  int total = (EE + NN) * (HD / 4);
  if (t >= total) return;
  int q = t & 15;          // which float4 of the 64-wide row
  int e = t >> 4;
  int rr = (e < EE) ? ei[e] : (e - EE);
  int cc = (e < EE) ? ei[EE + e] : (e - EE);
  float nrm = norm[e];
  float4 mv = *(const float4*)(Msum + (size_t)rr * HD + q * 4);
  float* sp = sym + (size_t)cc * HD + q * 4;
  atomicAdd(sp + 0, mv.x * nrm);
  atomicAdd(sp + 1, mv.y * nrm);
  atomicAdd(sp + 2, mv.z * nrm);
  atomicAdd(sp + 3, mv.w * nrm);
}

__global__ void k_dH(const float* __restrict__ anti, const float* __restrict__ sym,
                     float* __restrict__ dH, _Float16* __restrict__ dHh) {
  int i = blockIdx.x * 256 + threadIdx.x;
  if (i < NN * HD) {
    float v = fmaxf(tanhf(anti[i] + sym[i]), 0.0f);
    dH[i]  = v;
    dHh[i] = (_Float16)v;
  }
}

__global__ void k_updateH(float* __restrict__ Hm, const float* __restrict__ dH,
                          _Float16* __restrict__ Hh) {
  int i = blockIdx.x * 256 + threadIdx.x;
  if (i < NN * HD) {
    float h = Hm[i] + dH[i];
    Hm[i] = h;
    Hh[i] = (_Float16)h;
  }
}

// ---------------- pair MLP + gate + r update ----------------
// huv staged by async global->LDS B128 copies from the f16 H mirror (ASYNCcnt path).

__global__ void k_pairs(const _Float16* __restrict__ Hh, const _Float16* __restrict__ dHh,
                        const int* __restrict__ pairs, const float* __restrict__ gum,
                        float* __restrict__ r, float* __restrict__ ts,
                        const _Float16* __restrict__ fc1_pk,
                        const _Float16* __restrict__ fc2_pk,
                        const _Float16* __restrict__ fv_pk) {
  __shared__ __align__(16) _Float16 sh_huv[4][16 * 128];
  __shared__ __align__(16) _Float16 sh_Z[4][16 * 128];
  __shared__ float sh_sc[4][16 * 4];   // logit0, logit1, nuRaw, tau
  int wid = threadIdx.x >> 5, lane = threadIdx.x & 31;
  int pt = blockIdx.x * 4 + wid;
  bool act = pt < (PP / 16);
  int p0 = pt * 16;

  if (act) {
    // 16 rows x 256B (src 128B | dst 128B), 16B chunks, 8 async copies/lane
    unsigned lds_base = (unsigned)(size_t)(&sh_huv[wid][0]);
#pragma unroll
    for (int j = 0; j < 8; ++j) {
      int chunk = lane + 32 * j;          // 0..255
      int m  = chunk >> 4;
      int cc = chunk & 15;
      int node = (cc < 8) ? pairs[2 * (p0 + m)] : pairs[2 * (p0 + m) + 1];
      const char* gp = (const char*)Hh + (size_t)node * (HD * 2) + (size_t)(cc & 7) * 16;
      async_copy_b128(lds_base + chunk * 16, gp);
    }
  }
  async_wait0();
  __syncthreads();

  int n = lane & 15, m0 = (lane >> 4) * 8;
  if (act) {
#pragma unroll
    for (int nt = 0; nt < 8; ++nt) {         // Z = relu(huv @ W_fc1), 16x128
      v8f acc = {0.f, 0.f, 0.f, 0.f, 0.f, 0.f, 0.f, 0.f};
#pragma unroll
      for (int kt = 0; kt < 4; ++kt)
        acc = wmma16(lds_load_A(&sh_huv[wid][0], 128, kt * 32),
                     load_B_packed(fc1_pk, 4, kt, nt), acc);
#pragma unroll
      for (int i = 0; i < 8; ++i)
        sh_Z[wid][(m0 + i) * 128 + nt * 16 + n] = (_Float16)fmaxf(acc[i], 0.0f);
    }
  }
  __syncthreads();

  v8f accL = {0.f, 0.f, 0.f, 0.f, 0.f, 0.f, 0.f, 0.f};
  v8f accV = {0.f, 0.f, 0.f, 0.f, 0.f, 0.f, 0.f, 0.f};
  if (act) {
#pragma unroll
    for (int kt = 0; kt < 4; ++kt) {
      accL = wmma16(lds_load_A(&sh_Z[wid][0], 128, kt * 32),
                    load_B_packed(fc2_pk, 4, kt, 0), accL);   // logits (cols 0,1)
      accV = wmma16(lds_load_A(&sh_huv[wid][0], 128, kt * 32),
                    load_B_packed(fv_pk, 4, kt, 0), accV);    // nu raw (col 0)
    }
    if (n == 0) {
#pragma unroll
      for (int i = 0; i < 8; ++i) {
        sh_sc[wid][(m0 + i) * 4 + 0] = accL[i];
        sh_sc[wid][(m0 + i) * 4 + 2] = accV[i];
      }
    } else if (n == 1) {
#pragma unroll
      for (int i = 0; i < 8; ++i) sh_sc[wid][(m0 + i) * 4 + 1] = accL[i];
    }
  }
  __syncthreads();

  if (act && lane < 16) {
    float l0 = sh_sc[wid][lane * 4 + 0];
    float l1 = sh_sc[wid][lane * 4 + 1];
    float v  = sh_sc[wid][lane * 4 + 2];
    float nu = ((v > 20.0f) ? v : log1pf(expf(v))) + 1.0f;      // softplus + NU0
    float g0 = gum[(size_t)(p0 + lane) * 2 + 0];
    float g1 = gum[(size_t)(p0 + lane) * 2 + 1];
    float tau = 1.0f / (1.0f + expf((l1 + g1 - l0 - g0) / nu)); // softmax col 0
    sh_sc[wid][lane * 4 + 3] = tau;
    ts[p0 + lane] += tau;
  }
  __syncthreads();

  if (act) {
    for (int idx = lane; idx < 16 * 128; idx += 32) {
      int m = idx >> 7, f = idx & 127;
      int node = (f < 64) ? pairs[2 * (p0 + m)] : pairs[2 * (p0 + m) + 1];
      r[(size_t)(p0 + m) * 128 + f] +=
          sh_sc[wid][m * 4 + 3] * (float)dHh[(size_t)node * HD + (f & 63)];
    }
  }
}

// ---------------- readout: scores = (relu(r@W_p1)@W_p2)[:,0]; emit ts ----------------

__global__ void k_scores(const float* __restrict__ r, const float* __restrict__ ts,
                         const _Float16* __restrict__ p1_pk,
                         const _Float16* __restrict__ p2_pk, float* __restrict__ out) {
  __shared__ __align__(16) _Float16 sh_r[4][16 * 128];
  __shared__ __align__(16) _Float16 sh_Y[4][16 * 64];
  __shared__ float sh_sc[4][16];
  int wid = threadIdx.x >> 5, lane = threadIdx.x & 31;
  int pt = blockIdx.x * 4 + wid;
  bool act = pt < (PP / 16);
  int p0 = pt * 16;
  if (act) {
    for (int idx = lane; idx < 16 * 128; idx += 32)
      sh_r[wid][idx] = (_Float16)r[(size_t)p0 * 128 + idx];
  }
  __syncthreads();
  int n = lane & 15, m0 = (lane >> 4) * 8;
  if (act) {
#pragma unroll
    for (int nt = 0; nt < 4; ++nt) {
      v8f acc = {0.f, 0.f, 0.f, 0.f, 0.f, 0.f, 0.f, 0.f};
#pragma unroll
      for (int kt = 0; kt < 4; ++kt)
        acc = wmma16(lds_load_A(&sh_r[wid][0], 128, kt * 32),
                     load_B_packed(p1_pk, 4, kt, nt), acc);
#pragma unroll
      for (int i = 0; i < 8; ++i)
        sh_Y[wid][(m0 + i) * 64 + nt * 16 + n] = (_Float16)fmaxf(acc[i], 0.0f);
    }
  }
  __syncthreads();
  if (act) {
    v8f acc = {0.f, 0.f, 0.f, 0.f, 0.f, 0.f, 0.f, 0.f};
#pragma unroll
    for (int kt = 0; kt < 2; ++kt)
      acc = wmma16(lds_load_A(&sh_Y[wid][0], 64, kt * 32),
                   load_B_packed(p2_pk, 2, kt, 0), acc);
    if (n == 0) {
#pragma unroll
      for (int i = 0; i < 8; ++i) sh_sc[wid][m0 + i] = acc[i];
    }
  }
  __syncthreads();
  if (act && lane < 16) {
    out[p0 + lane]      = sh_sc[wid][lane];
    out[PP + p0 + lane] = ts[p0 + lane];
  }
}

// ---------------- host orchestration ----------------

extern "C" void kernel_launch(void* const* d_in, const int* in_sizes, int n_in,
                              void* d_out, int out_size, void* d_ws, size_t ws_size,
                              hipStream_t stream) {
  const float* x     = (const float*)d_in[0];
  const int*   ei    = (const int*)d_in[1];
  const int*   pairs = (const int*)d_in[2];
  const float* gum   = (const float*)d_in[3];
  const float* Wenc  = (const float*)d_in[4];
  const float* Omega = (const float*)d_in[5];
  const float* WsRaw = (const float*)d_in[6];
  const float* Wfc1  = (const float*)d_in[7];
  const float* Wfc2  = (const float*)d_in[8];
  const float* Wfv   = (const float*)d_in[9];
  const float* Wp1   = (const float*)d_in[10];
  const float* Wp2   = (const float*)d_in[11];
  float* out = (float*)d_out;

  size_t cur = 0;
  auto carve = [&](size_t bytes) -> void* {
    void* p = (char*)d_ws + cur;
    cur = (cur + bytes + 255) & ~(size_t)255;
    return p;
  };
  float* deg     = (float*)carve(sizeof(float) * NN);
  float* dinv    = (float*)carve(sizeof(float) * NN);
  float* norm    = (float*)carve(sizeof(float) * (EE + NN));
  float* Hm      = (float*)carve(sizeof(float) * NN * HD);
  float* anti    = (float*)carve(sizeof(float) * NN * HD);
  float* Msum    = (float*)carve(sizeof(float) * NN * HD);
  float* sym     = (float*)carve(sizeof(float) * NN * HD);
  float* dHb     = (float*)carve(sizeof(float) * NN * HD);
  float* rb      = (float*)carve(sizeof(float) * (size_t)PP * 128);
  float* tsb     = (float*)carve(sizeof(float) * PP);
  float* OmegaAs = (float*)carve(sizeof(float) * HD * HD);
  float* Wsym    = (float*)carve(sizeof(float) * HD * HD);
  float* Wfc2pad = (float*)carve(sizeof(float) * 128 * 16);
  float* Wfvpad  = (float*)carve(sizeof(float) * 128 * 16);
  float* Wp2pad  = (float*)carve(sizeof(float) * 64 * 16);
  _Float16* Hh     = (_Float16*)carve(2 * NN * HD);
  _Float16* dHh    = (_Float16*)carve(2 * NN * HD);
  _Float16* enc_pk = (_Float16*)carve(2 * 128 * 64);
  _Float16* om_pk  = (_Float16*)carve(2 * 64 * 64);
  _Float16* ws_pk  = (_Float16*)carve(2 * 64 * 64);
  _Float16* fc1_pk = (_Float16*)carve(2 * 128 * 128);
  _Float16* fc2_pk = (_Float16*)carve(2 * 128 * 16);
  _Float16* fv_pk  = (_Float16*)carve(2 * 128 * 16);
  _Float16* p1_pk  = (_Float16*)carve(2 * 128 * 64);
  _Float16* p2_pk  = (_Float16*)carve(2 * 64 * 16);

  auto cdiv = [](long long a, long long b) { return (int)((a + b - 1) / b); };

  // weight prep + packing (tiny, L2-resident)
  k_prep_small<<<cdiv(64 * 64, 256), 256, 0, stream>>>(Omega, WsRaw, Wfc2, Wfv, Wp2,
                                                       OmegaAs, Wsym, Wfc2pad, Wfvpad, Wp2pad);
  k_pack_weight<<<cdiv(128 * 64, 256), 256, 0, stream>>>(Wenc, enc_pk, 128, 64);
  k_pack_weight<<<cdiv(64 * 64, 256), 256, 0, stream>>>(OmegaAs, om_pk, 64, 64);
  k_pack_weight<<<cdiv(64 * 64, 256), 256, 0, stream>>>(Wsym, ws_pk, 64, 64);
  k_pack_weight<<<cdiv(128 * 128, 256), 256, 0, stream>>>(Wfc1, fc1_pk, 128, 128);
  k_pack_weight<<<cdiv(128 * 16, 256), 256, 0, stream>>>(Wfc2pad, fc2_pk, 128, 16);
  k_pack_weight<<<cdiv(128 * 16, 256), 256, 0, stream>>>(Wfvpad, fv_pk, 128, 16);
  k_pack_weight<<<cdiv(128 * 64, 256), 256, 0, stream>>>(Wp1, p1_pk, 128, 64);
  k_pack_weight<<<cdiv(64 * 16, 256), 256, 0, stream>>>(Wp2pad, p2_pk, 64, 16);

  // graph normalization
  k_init<<<cdiv(PP, 256), 256, 0, stream>>>(deg, tsb);
  k_deg<<<cdiv(EE, 256), 256, 0, stream>>>(ei, deg);
  k_dinv<<<cdiv(NN, 256), 256, 0, stream>>>(deg, dinv);
  k_norm<<<cdiv(EE + NN, 256), 256, 0, stream>>>(ei, dinv, norm);

  // encoder + r init
  k_encode<<<cdiv(NN / 16, 4), 128, 0, stream>>>(x, enc_pk, Hm, Hh);
  k_rinit<<<cdiv((long long)PP * 128, 256), 256, 0, stream>>>(Hm, pairs, rb);

  // 20 scan steps
  for (int t = 0; t < LL; ++t) {
    k_HW<<<cdiv(NN / 16, 8), 256, 0, stream>>>(Hm, om_pk, ws_pk, anti, Msum);
    k_zero<<<cdiv((long long)NN * HD, 256), 256, 0, stream>>>(sym, (long long)NN * HD);
    k_scatter<<<cdiv((long long)(EE + NN) * (HD / 4), 256), 256, 0, stream>>>(ei, norm, Msum, sym);
    k_dH<<<cdiv(NN * HD, 256), 256, 0, stream>>>(anti, sym, dHb, dHh);
    k_pairs<<<cdiv(PP / 16, 4), 128, 0, stream>>>(Hh, dHh, pairs,
                                                  gum + (size_t)t * PP * 2, rb, tsb,
                                                  fc1_pk, fc2_pk, fv_pk);
    k_updateH<<<cdiv(NN * HD, 256), 256, 0, stream>>>(Hm, dHb, Hh);
  }

  // readout
  k_scores<<<cdiv(PP / 16, 4), 128, 0, stream>>>(rb, tsb, p1_pk, p2_pk, out);
}
